// Net_33998961115614
// MI455X (gfx1250) — matrix operand order repaired and checked by reading, not compiled, not run on hardware
//
#include <hip/hip_runtime.h>
#include <hip/hip_bf16.h>
#include <stdint.h>

// ---- model dims (match reference) ----
#define N_NODES   50000
#define N_EDGES   600000
#define EP_EDGES  (N_EDGES + N_NODES)   // with self loops
#define F_IN      256
#define HID       128
#define NHID      64
#define NCLASS    2
#define NGRAPH    128
#define NEG_SLOPE 0.2f

typedef __attribute__((ext_vector_type(16))) _Float16 v16h;
typedef __attribute__((ext_vector_type(8)))  _Float16 v8h;
typedef __attribute__((ext_vector_type(8)))  float    v8f;

// ---------------- helpers ----------------
__device__ __forceinline__ float selu_f(float v) {
  const float scale = 1.0507009873554805f;
  const float alpha = 1.6732632423543772f;
  return scale * (v > 0.f ? v : alpha * (__expf(v) - 1.f));
}

__device__ __forceinline__ void atomicMaxFloat(float* addr, float val) {
  // works when addr initialized to -inf (0xFF800000)
  if (val >= 0.f) atomicMax((int*)addr, __float_as_int(val));
  else            atomicMin((unsigned int*)addr, __float_as_uint(val));
}

// ---------------- fill ----------------
__global__ void fill_u32_kernel(uint32_t* __restrict__ p, uint32_t v, size_t n) {
  size_t i = (size_t)blockIdx.x * blockDim.x + threadIdx.x;
  if (i < n) p[i] = v;
}

// ---------------- GEMM: C[M x 128] = A[M x K] * W[K x 128] via WMMA f16 ----------------
// One wave per 16x16 output tile; 8 waves/block cover all 128 output columns.
// W is staged once per block into LDS, transposed ([col][k]) and converted to f16,
// so the B operand is two contiguous ds_load_b128 per K-step per lane.
// A is read as four float4 (global_load_b128) per K-step per lane (rows are full:
// M % 16 == 0 for this model, so no predication in the hot loop -> EXEC all ones).
__global__ void gemm_wmma_kernel(const float* __restrict__ A, const float* __restrict__ W,
                                 float* __restrict__ C, int M, int K) {
  extern __shared__ _Float16 Wlds[];          // [HID][K+8] halves, padded stride
  const int LDW = K + 8;                      // pad kills bank conflicts on staging

  // ---- stage W (K x 128, row major) -> Wlds transposed as [col][k], f16 ----
  for (int i = threadIdx.x; i < K * HID; i += blockDim.x) {
    int k = i >> 7;            // row of W
    int c = i & (HID - 1);     // col of W (coalesced global read)
    Wlds[c * LDW + k] = (_Float16)W[i];
  }
  __syncthreads();

  const int lane    = threadIdx.x & 31;
  const int wave    = threadIdx.x >> 5;       // 0..7 -> column tile
  const int rowBase = blockIdx.x * 16;
  const int colBase = wave * 16;
  const int hi      = (lane >> 4);            // 0: lanes 0-15, 1: lanes 16-31
  const int l15     = lane & 15;

  v8f acc = {};
  const float*    Arow = A + (size_t)(rowBase + l15) * K;   // M%16==0: always valid
  const float4*   Arow4 = (const float4*)Arow;
  const _Float16* wcol  = Wlds + (colBase + l15) * LDW;     // 16B-aligned (LDW*2 % 16 == 0)

  for (int kk = 0; kk < K; kk += 32) {
    if (kk + 32 < K) __builtin_prefetch(Arow + kk + 32, 0, 0);  // global_prefetch_b8

    // A operand (ISA 7.12.2, 16-bit A 16x32): lane's 16 elems are two contiguous
    // 8-float runs: [kk + hi*8, +8) and [kk + 16 + hi*8, +8)
    const int q0 = (kk + hi * 8) >> 2;
    float4 a0 = Arow4[q0];
    float4 a1 = Arow4[q0 + 1];
    float4 a2 = Arow4[q0 + 4];
    float4 a3 = Arow4[q0 + 5];
    v16h a;
    a[0]  = (_Float16)a0.x;  a[1]  = (_Float16)a0.y;
    a[2]  = (_Float16)a0.z;  a[3]  = (_Float16)a0.w;
    a[4]  = (_Float16)a1.x;  a[5]  = (_Float16)a1.y;
    a[6]  = (_Float16)a1.z;  a[7]  = (_Float16)a1.w;
    a[8]  = (_Float16)a2.x;  a[9]  = (_Float16)a2.y;
    a[10] = (_Float16)a2.z;  a[11] = (_Float16)a2.w;
    a[12] = (_Float16)a3.x;  a[13] = (_Float16)a3.y;
    a[14] = (_Float16)a3.z;  a[15] = (_Float16)a3.w;

    // B operand (32x16, lane = column): two 8-half runs down W's column: [kk + hi*8, +8) and [kk + 16 + hi*8, +8)
    const _Float16* wp = wcol + kk + hi * 8;
    v8h b0 = *(const v8h*)(wp);
    v8h b1 = *(const v8h*)(wp + 16);
    v16h b;
#pragma unroll
    for (int j = 0; j < 8; ++j) { b[j] = b0[j]; b[8 + j] = b1[j]; }

    acc = __builtin_amdgcn_wmma_f32_16x16x32_f16(false, a, false, b,
                                                 (short)0, acc, false, false);
  }

  // C/D layout: VGPR r -> row rowBase + r (+8 for upper half lanes), col = lane&15
#pragma unroll
  for (int r = 0; r < 8; ++r) {
    int row = rowBase + r + (hi ? 8 : 0);
    if (row < M) C[(size_t)row * HID + colBase + l15] = acc[r];
  }
}

// ---------------- per-node attention scalars: asrc = h.a_src, adst = h.a_dst ----------------
__global__ void attn_kernel(const float* __restrict__ h, const float* __restrict__ a_src,
                            const float* __restrict__ a_dst, float* __restrict__ asrc,
                            float* __restrict__ adst, int n) {
  int lane = threadIdx.x & 31;
  int node = blockIdx.x * (blockDim.x >> 5) + (threadIdx.x >> 5);
  if (node >= n) return;
  const float* hp = h + (size_t)node * HID;
  float s = 0.f, d = 0.f;
#pragma unroll
  for (int i = 0; i < HID / 32; ++i) {
    int f = lane + 32 * i;
    float v = hp[f];
    s += v * a_src[f];
    d += v * a_dst[f];
  }
#pragma unroll
  for (int off = 16; off > 0; off >>= 1) {
    s += __shfl_down(s, off, 32);
    d += __shfl_down(d, off, 32);
  }
  if (lane == 0) { asrc[node] = s; adst[node] = d; }
}

// ---------------- edge pass 1: segment max of leaky_relu(asrc[s]+adst[d]) ----------------
__global__ void edge_max_kernel(const int* __restrict__ ei,
                                const float* __restrict__ asrc, const float* __restrict__ adst,
                                float* __restrict__ emax) {
  int e = blockIdx.x * blockDim.x + threadIdx.x;
  if (e >= EP_EDGES) return;
  int s = (e < N_EDGES) ? ei[e]           : (e - N_EDGES);
  int d = (e < N_EDGES) ? ei[N_EDGES + e] : (e - N_EDGES);
  float v = asrc[s] + adst[d];
  v = (v > 0.f) ? v : NEG_SLOPE * v;
  atomicMaxFloat(&emax[d], v);
}

// ---------------- edge pass 2: ex = exp(e - emax[d]); denom[d] += ex ----------------
__global__ void edge_exp_kernel(const int* __restrict__ ei,
                                const float* __restrict__ asrc, const float* __restrict__ adst,
                                const float* __restrict__ emax,
                                float* __restrict__ ex, float* __restrict__ denom) {
  int e = blockIdx.x * blockDim.x + threadIdx.x;
  if (e >= EP_EDGES) return;
  int s = (e < N_EDGES) ? ei[e]           : (e - N_EDGES);
  int d = (e < N_EDGES) ? ei[N_EDGES + e] : (e - N_EDGES);
  float v = asrc[s] + adst[d];
  v = (v > 0.f) ? v : NEG_SLOPE * v;
  float t = __expf(v - emax[d]);
  ex[e] = t;
  atomicAdd(&denom[d], t);
}

// ---------------- edge pass 3: agg[d] += h[s] * (ex/denom[d]) ; wave per edge ----------------
__global__ void edge_agg_kernel(const int* __restrict__ ei, const float* __restrict__ h,
                                const float* __restrict__ ex, const float* __restrict__ denom,
                                float* __restrict__ agg) {
  int e    = blockIdx.x * (blockDim.x >> 5) + (threadIdx.x >> 5);
  int lane = threadIdx.x & 31;
  if (e >= EP_EDGES) return;
  int s = (e < N_EDGES) ? ei[e]           : (e - N_EDGES);
  int d = (e < N_EDGES) ? ei[N_EDGES + e] : (e - N_EDGES);
  float coef = ex[e] / denom[d];
  const float* hs = h   + (size_t)s * HID;
  float*       ad = agg + (size_t)d * HID;
#pragma unroll
  for (int i = 0; i < HID / 32; ++i) {
    int f = lane + 32 * i;
    atomicAdd(&ad[f], hs[f] * coef);
  }
}

// ---------------- out = selu(agg + bias) ----------------
__global__ void bias_selu_kernel(const float* __restrict__ agg, const float* __restrict__ b,
                                 float* __restrict__ out, int n) {
  size_t i = (size_t)blockIdx.x * blockDim.x + threadIdx.x;
  if (i >= (size_t)n * HID) return;
  int f = (int)(i & (HID - 1));
  out[i] = selu_f(agg[i] + b[f]);
}

// ---------------- global mean pool (sum + counts) ----------------
__global__ void pool_kernel(const float* __restrict__ emb, const int* __restrict__ batch,
                            float* __restrict__ pooled, float* __restrict__ counts) {
  size_t i = (size_t)blockIdx.x * blockDim.x + threadIdx.x;
  if (i >= (size_t)N_NODES * HID) return;
  int node = (int)(i >> 7);
  int f    = (int)(i & (HID - 1));
  int g    = batch[node];
  atomicAdd(&pooled[(size_t)g * HID + f], emb[i]);
  if (f == 0) atomicAdd(&counts[g], 1.0f);
}

// ---------------- head: mean, selu, fc1+selu, fc2, log_softmax ----------------
__global__ void head_kernel(const float* __restrict__ pooled, const float* __restrict__ counts,
                            const float* __restrict__ fc1w, const float* __restrict__ fc1b,
                            const float* __restrict__ fc2w, const float* __restrict__ fc2b,
                            float* __restrict__ out /* [NGRAPH x NCLASS] */) {
  __shared__ float z[HID];
  __shared__ float z1[NHID];
  __shared__ float lg[NCLASS];
  int g = blockIdx.x;
  int t = threadIdx.x;
  float c = counts[g];
  c = c > 1.f ? c : 1.f;
  if (t < HID) z[t] = selu_f(pooled[(size_t)g * HID + t] / c);
  __syncthreads();
  if (t < NHID) {
    float acc = fc1b[t];
    for (int k = 0; k < HID; ++k) acc += z[k] * fc1w[k * NHID + t];
    z1[t] = selu_f(acc);
  }
  __syncthreads();
  if (t < NCLASS) {
    float acc = fc2b[t];
    for (int k = 0; k < NHID; ++k) acc += z1[k] * fc2w[k * NCLASS + t];
    lg[t] = acc;
  }
  __syncthreads();
  if (t < NCLASS) {
    float m = fmaxf(lg[0], lg[1]);
    float lse = m + __logf(__expf(lg[0] - m) + __expf(lg[1] - m));
    out[(size_t)g * NCLASS + t] = lg[t] - lse;
  }
}

// =====================================================================
extern "C" void kernel_launch(void* const* d_in, const int* in_sizes, int n_in,
                              void* d_out, int out_size, void* d_ws, size_t ws_size,
                              hipStream_t stream) {
  (void)in_sizes; (void)n_in; (void)out_size; (void)ws_size;

  const float* x      = (const float*)d_in[0];
  const int*   ei     = (const int*)  d_in[1];   // [2 x E]
  const int*   batch  = (const int*)  d_in[2];
  const float* W1     = (const float*)d_in[3];
  const float* a_src1 = (const float*)d_in[4];
  const float* a_dst1 = (const float*)d_in[5];
  const float* b1     = (const float*)d_in[6];
  const float* W2     = (const float*)d_in[7];
  const float* a_src2 = (const float*)d_in[8];
  const float* a_dst2 = (const float*)d_in[9];
  const float* b2     = (const float*)d_in[10];
  const float* fc1w   = (const float*)d_in[11];
  const float* fc1b   = (const float*)d_in[12];
  const float* fc2w   = (const float*)d_in[13];
  const float* fc2b   = (const float*)d_in[14];

  // workspace layout (floats)
  float* ws     = (float*)d_ws;
  float* bufA   = ws;                                  // N*HID  : GEMM output h
  float* bufB   = bufA  + (size_t)N_NODES * HID;       // N*HID  : aggregation / layer out
  float* asrc   = bufB  + (size_t)N_NODES * HID;       // N
  float* adst   = asrc  + N_NODES;                     // N
  float* emax   = adst  + N_NODES;                     // N
  float* denom  = emax  + N_NODES;                     // N
  float* exbuf  = denom + N_NODES;                     // EP
  float* pooled = exbuf + EP_EDGES;                    // G*HID
  float* counts = pooled + (size_t)NGRAPH * HID;       // G

  float* out_emb  = (float*)d_out;                     // [N x HID]
  float* out_logp = out_emb + (size_t)N_NODES * HID;   // [G x NCLASS]

  const int TB = 256;
  dim3 blk(TB);
  auto cdiv = [](size_t a, size_t b) { return (unsigned)((a + b - 1) / b); };

  const unsigned gFillN   = cdiv(N_NODES, TB);
  const unsigned gFillNH  = cdiv((size_t)N_NODES * HID, TB);
  const unsigned gEdge    = cdiv(EP_EDGES, TB);
  const unsigned gEdgeW   = cdiv(EP_EDGES, TB / 32);   // wave per edge
  const unsigned gAttn    = cdiv(N_NODES, TB / 32);
  const unsigned gGemm    = cdiv(N_NODES, 16);

  const size_t shW1 = (size_t)HID * (F_IN + 8) * sizeof(_Float16);  // ~66 KB
  const size_t shW2 = (size_t)HID * (HID  + 8) * sizeof(_Float16);  // ~34 KB

  // ===================== layer 1 =====================
  gemm_wmma_kernel<<<gGemm, blk, shW1, stream>>>(x, W1, bufA, N_NODES, F_IN);
  attn_kernel<<<gAttn, blk, 0, stream>>>(bufA, a_src1, a_dst1, asrc, adst, N_NODES);

  fill_u32_kernel<<<gFillN,  blk, 0, stream>>>((uint32_t*)emax,  0xFF800000u, N_NODES); // -inf
  fill_u32_kernel<<<gFillN,  blk, 0, stream>>>((uint32_t*)denom, 0u, N_NODES);
  fill_u32_kernel<<<gFillNH, blk, 0, stream>>>((uint32_t*)bufB,  0u, (size_t)N_NODES * HID);

  edge_max_kernel<<<gEdge, blk, 0, stream>>>(ei, asrc, adst, emax);
  edge_exp_kernel<<<gEdge, blk, 0, stream>>>(ei, asrc, adst, emax, exbuf, denom);
  edge_agg_kernel<<<gEdgeW, blk, 0, stream>>>(ei, bufA, exbuf, denom, bufB);
  bias_selu_kernel<<<gFillNH, blk, 0, stream>>>(bufB, b1, bufB, N_NODES);   // in-place

  // ===================== layer 2 =====================
  gemm_wmma_kernel<<<gGemm, blk, shW2, stream>>>(bufB, W2, bufA, N_NODES, HID);
  attn_kernel<<<gAttn, blk, 0, stream>>>(bufA, a_src2, a_dst2, asrc, adst, N_NODES);

  fill_u32_kernel<<<gFillN,  blk, 0, stream>>>((uint32_t*)emax,  0xFF800000u, N_NODES);
  fill_u32_kernel<<<gFillN,  blk, 0, stream>>>((uint32_t*)denom, 0u, N_NODES);
  fill_u32_kernel<<<gFillNH, blk, 0, stream>>>((uint32_t*)bufB,  0u, (size_t)N_NODES * HID);

  edge_max_kernel<<<gEdge, blk, 0, stream>>>(ei, asrc, adst, emax);
  edge_exp_kernel<<<gEdge, blk, 0, stream>>>(ei, asrc, adst, emax, exbuf, denom);
  edge_agg_kernel<<<gEdgeW, blk, 0, stream>>>(ei, bufA, exbuf, denom, bufB);
  bias_selu_kernel<<<gFillNH, blk, 0, stream>>>(bufB, b2, out_emb, N_NODES); // embedding out

  // ===================== pooling + head =====================
  fill_u32_kernel<<<cdiv((size_t)NGRAPH * HID, TB), blk, 0, stream>>>((uint32_t*)pooled, 0u,
                                                                      (size_t)NGRAPH * HID);
  fill_u32_kernel<<<1, blk, 0, stream>>>((uint32_t*)counts, 0u, NGRAPH);
  pool_kernel<<<gFillNH, blk, 0, stream>>>(out_emb, batch, pooled, counts);
  head_kernel<<<NGRAPH, dim3(HID), 0, stream>>>(pooled, counts, fc1w, fc1b, fc2w, fc2b, out_logp);
}
